// HGTLayer_21492016349919
// MI455X (gfx1250) — matrix operand order repaired and checked by reading, not compiled, not run on hardware
//
#include <hip/hip_runtime.h>
#include <stdint.h>

#define DIM    256
#define NHEAD  8
#define HDIM   32

// ---------- CDNA5 WMMA types ----------
typedef __attribute__((ext_vector_type(16))) __bf16 v16bf;
typedef __attribute__((ext_vector_type(8)))  float  v8f;

union Frag16 {
    v16bf v;
    uint4 q[2];
    unsigned short h[16];
};

__device__ __forceinline__ unsigned short f2bf(float f) {
    union { float f; unsigned int u; } c; c.f = f;
    unsigned int u = c.u;
    u += 0x7FFFu + ((u >> 16) & 1u);   // round-to-nearest-even
    return (unsigned short)(u >> 16);
}

__device__ __forceinline__ void wait_asynccnt0() {
#if __has_builtin(__builtin_amdgcn_s_wait_asynccnt)
    __builtin_amdgcn_s_wait_asynccnt(0);
#else
    asm volatile("s_wait_asynccnt 0x0" ::: "memory");
#endif
}

// ---------- f32 -> bf16 conversion ----------
__global__ __launch_bounds__(256)
void cvt_bf16_kernel(const float* __restrict__ src, unsigned short* __restrict__ dst, size_t n) {
    size_t i = (size_t)blockIdx.x * blockDim.x + threadIdx.x;
    size_t stride = (size_t)gridDim.x * blockDim.x;
    for (; i < n; i += stride) dst[i] = f2bf(src[i]);
}

// ---------- Wq_eff[c, h*32+f] = sum_d Wq[c, h*32+d] * R[h,d,f]  (c==256 row => bias) ----------
__global__ __launch_bounds__(256)
void wqeff_kernel(const float* __restrict__ Wq, const float* __restrict__ bq,
                  const float* __restrict__ R, float* __restrict__ Weff,
                  float* __restrict__ beff) {
    int idx = blockIdx.x * blockDim.x + threadIdx.x;   // 257*256 total
    if (idx >= 257 * 256) return;
    int o = idx & 255;
    int c = idx >> 8;          // 0..256; 256 == bias row
    int h = o >> 5, f = o & 31;
    const float* Rh = R + (size_t)h * HDIM * HDIM;     // R[h][d][f] = Rh[d*32+f]
    float acc = 0.f;
    if (c < 256) {
        const float* wrow = Wq + (size_t)c * DIM + h * HDIM;
#pragma unroll
        for (int d = 0; d < HDIM; ++d) acc += wrow[d] * Rh[d * HDIM + f];
        Weff[(size_t)c * DIM + o] = acc;
    } else {
        const float* brow = bq + h * HDIM;
#pragma unroll
        for (int d = 0; d < HDIM; ++d) acc += brow[d] * Rh[d * HDIM + f];
        beff[o] = acc;
    }
}

// ---------- pack 256x256 f32 W into bf16 WMMA B-fragment order ----------
// packed[((tn*8+ks)*32+lane)*16 + i] = W[ks*32 + 16*(lane>>4) + i][tn*16 + (lane&15)]
__global__ __launch_bounds__(256)
void pack_b_kernel(const float* __restrict__ W, unsigned short* __restrict__ P) {
    int idx = blockIdx.x * blockDim.x + threadIdx.x;   // 65536 total
    if (idx >= 65536) return;
    int i    = idx & 15;
    int lane = (idx >> 4) & 31;
    int ks   = (idx >> 9) & 7;
    int tn   = idx >> 12;
    int g = lane >> 4, n = lane & 15;
    int k = ks * 32 + 16 * g + i;
    int c = tn * 16 + n;
    P[idx] = f2bf(W[(size_t)k * DIM + c]);
}

// ---------- WMMA GEMM: Y = Xb(bf16) @ Wpacked(bf16) + bias, f32 out ----------
// block = 256 threads = 8 waves; block owns 16 rows x 256 cols; wave w -> n-tiles {2w, 2w+1}
// A tile (16x256 bf16, 8KB) staged to LDS once per block via async-to-LDS DMA.
__global__ __launch_bounds__(256)
void gemm_bias_kernel(const unsigned short* __restrict__ Xb,
                      const unsigned short* __restrict__ Wp,
                      const float* __restrict__ bias,
                      float* __restrict__ Y, int M) {
    __shared__ unsigned short atile[16 * DIM];   // 8 KB

    const int m0 = blockIdx.x << 4;
    if (m0 >= M) return;                         // uniform per block

    // ---- stage A tile: 256 threads x 32B, coalesced row segments ----
    {
        const int t   = threadIdx.x;
        const int row = t >> 4;
        const int seg = t & 15;                  // 16 elements (32B) per thread
        const unsigned short* gp = Xb + (size_t)(m0 + row) * DIM + seg * 16;
        const unsigned long long g64 = (unsigned long long)(uintptr_t)gp;
        const unsigned lds0 = (unsigned)(uintptr_t)(atile + row * DIM + seg * 16);
        // INST_OFFSET is added to both LDS and global addresses (ISA 08 §4.4)
        asm volatile("global_load_async_to_lds_b128 %0, %1, off\n\t"
                     "global_load_async_to_lds_b128 %0, %1, off offset:16"
                     :: "v"(lds0), "v"(g64) : "memory");
    }
    wait_asynccnt0();
    __syncthreads();

    const int lane = threadIdx.x & 31;
    const int wave = threadIdx.x >> 5;
    const int g  = lane >> 4;
    const int mn = lane & 15;
    const int tn0 = wave * 2, tn1 = tn0 + 1;

    const unsigned short* arow = atile + mn * DIM;   // LDS reads -> ds_load_b128
    v8f acc0 = {}; v8f acc1 = {};
#pragma unroll
    for (int ks = 0; ks < 8; ++ks) {
        Frag16 a, b0, b1;
        const unsigned short* ap = arow + ks * 32 + 16 * g;   // two 16B chunks per lane
        a.q[0] = *(const uint4*)(ap);
        a.q[1] = *(const uint4*)(ap + 8);
        const unsigned short* bp0 = Wp + (((size_t)(tn0 * 8 + ks) * 32 + lane) << 4);
        b0.q[0] = *(const uint4*)(bp0); b0.q[1] = *(const uint4*)(bp0 + 8);
        const unsigned short* bp1 = Wp + (((size_t)(tn1 * 8 + ks) * 32 + lane) << 4);
        b1.q[0] = *(const uint4*)(bp1); b1.q[1] = *(const uint4*)(bp1 + 8);
        acc0 = __builtin_amdgcn_wmma_f32_16x16x32_bf16(false, a.v, false, b0.v, (short)0, acc0, false, false);
        acc1 = __builtin_amdgcn_wmma_f32_16x16x32_bf16(false, a.v, false, b1.v, (short)0, acc1, false, false);
    }
    const float bc0 = bias[tn0 * 16 + mn];
    const float bc1 = bias[tn1 * 16 + mn];
    const size_t rb = (size_t)(m0 + 8 * g);
#pragma unroll
    for (int r = 0; r < 8; ++r) {
        Y[(rb + r) * DIM + tn0 * 16 + mn] = acc0[r] + bc0;
        Y[(rb + r) * DIM + tn1 * 16 + mn] = acc1[r] + bc1;
    }
}

// ---------- edge message: one wave32 per edge ----------
__global__ __launch_bounds__(256)
void edge_kernel(const float* __restrict__ qW, const float* __restrict__ K,
                 const float* __restrict__ V, const int* __restrict__ ei, int E,
                 float* __restrict__ accum, float* __restrict__ norm) {
    const int lane = threadIdx.x & 31;
    const int e = (blockIdx.x * blockDim.x + threadIdx.x) >> 5;
    if (e >= E) return;
    const int src = ei[e];
    const int dst = ei[E + e];
    const float* q = qW + (size_t)dst * DIM;
    const float* k = K  + (size_t)src * DIM;
    const float* v = V  + (size_t)src * DIM;
    float* ac = accum + (size_t)dst * DIM;
    const float inv_scale = 0.17677669529663687f;   // 1/sqrt(32)
    float asum = 0.f;
#pragma unroll
    for (int h = 0; h < NHEAD; ++h) {
        const int c = h * HDIM + lane;
        float p = q[c] * k[c];
        p += __shfl_xor(p, 16); p += __shfl_xor(p, 8); p += __shfl_xor(p, 4);
        p += __shfl_xor(p, 2);  p += __shfl_xor(p, 1);
        float s = fminf(fmaxf(p * inv_scale, -5.f), 5.f);
        float ae = __expf(s);
        asum += ae;
        atomicAdd(&ac[c], v[c] * ae);
    }
    if (lane == 0) atomicAdd(&norm[dst], asum * 0.125f);
}

// ---------- output GEMM: Y = (accum/norm) @ Wo + bo + x_residual ----------
__global__ __launch_bounds__(256)
void gemm_out_kernel(const float* __restrict__ accum, const float* __restrict__ norm,
                     const unsigned short* __restrict__ Wp, const float* __restrict__ bo,
                     const float* __restrict__ xres, float* __restrict__ Y, int M) {
    const int lane = threadIdx.x & 31;
    const int wave = threadIdx.x >> 5;
    const int m0 = blockIdx.x << 4;
    if (m0 >= M) return;
    const int g  = lane >> 4;
    const int mn = lane & 15;
    const int tn0 = wave * 2, tn1 = tn0 + 1;
    const float inv = 1.f / fmaxf(norm[m0 + mn], 1e-8f);
    const float* ar = accum + (size_t)(m0 + mn) * DIM;
    v8f acc0 = {}; v8f acc1 = {};
#pragma unroll
    for (int ks = 0; ks < 8; ++ks) {
        Frag16 a, b0, b1;
        const float* ap = ar + ks * 32 + 16 * g;
#pragma unroll
        for (int i = 0; i < 8; ++i) a.h[i]     = f2bf(ap[i] * inv);
#pragma unroll
        for (int i = 0; i < 8; ++i) a.h[8 + i] = f2bf(ap[8 + i] * inv);
        const unsigned short* bp0 = Wp + (((size_t)(tn0 * 8 + ks) * 32 + lane) << 4);
        b0.q[0] = *(const uint4*)(bp0); b0.q[1] = *(const uint4*)(bp0 + 8);
        const unsigned short* bp1 = Wp + (((size_t)(tn1 * 8 + ks) * 32 + lane) << 4);
        b1.q[0] = *(const uint4*)(bp1); b1.q[1] = *(const uint4*)(bp1 + 8);
        acc0 = __builtin_amdgcn_wmma_f32_16x16x32_bf16(false, a.v, false, b0.v, (short)0, acc0, false, false);
        acc1 = __builtin_amdgcn_wmma_f32_16x16x32_bf16(false, a.v, false, b1.v, (short)0, acc1, false, false);
    }
    const size_t rb = (size_t)(m0 + 8 * g);
    const float bo0 = bo[tn0 * 16 + mn];
    const float bo1 = bo[tn1 * 16 + mn];
#pragma unroll
    for (int r = 0; r < 8; ++r) {
        size_t i0 = (rb + r) * DIM + tn0 * 16 + mn;
        size_t i1 = (rb + r) * DIM + tn1 * 16 + mn;
        Y[i0] = acc0[r] + bo0 + xres[i0];
        Y[i1] = acc1[r] + bo1 + xres[i1];
    }
}

// ---------- LayerNorm: one wave32 per row ----------
__global__ __launch_bounds__(256)
void ln_kernel(const float* __restrict__ Yin, const float* __restrict__ gain,
               const float* __restrict__ beta, float* __restrict__ out, int M) {
    const int lane = threadIdx.x & 31;
    const int row = (blockIdx.x * blockDim.x + threadIdx.x) >> 5;
    if (row >= M) return;
    const float* yr = Yin + (size_t)row * DIM;
    float v[8]; float s = 0.f;
#pragma unroll
    for (int j = 0; j < 8; ++j) { v[j] = yr[lane + 32 * j]; s += v[j]; }
    s += __shfl_xor(s, 16); s += __shfl_xor(s, 8); s += __shfl_xor(s, 4);
    s += __shfl_xor(s, 2);  s += __shfl_xor(s, 1);
    const float mean = s * (1.f / 256.f);
    float vs = 0.f;
#pragma unroll
    for (int j = 0; j < 8; ++j) { float d = v[j] - mean; vs += d * d; }
    vs += __shfl_xor(vs, 16); vs += __shfl_xor(vs, 8); vs += __shfl_xor(vs, 4);
    vs += __shfl_xor(vs, 2);  vs += __shfl_xor(vs, 1);
    const float rstd = rsqrtf(vs * (1.f / 256.f) + 1e-5f);
    float* orow = out + (size_t)row * DIM;
#pragma unroll
    for (int j = 0; j < 8; ++j) {
        const int c = lane + 32 * j;
        orow[c] = (v[j] - mean) * rstd * gain[c] + beta[c];
    }
}

extern "C" void kernel_launch(void* const* d_in, const int* in_sizes, int n_in,
                              void* d_out, int out_size, void* d_ws, size_t ws_size,
                              hipStream_t stream) {
    const float* x_p  = (const float*)d_in[0];
    const float* x_a  = (const float*)d_in[1];
    const float* Wk_p = (const float*)d_in[2];
    const float* bk_p = (const float*)d_in[3];
    const float* Wq_p = (const float*)d_in[4];
    const float* bq_p = (const float*)d_in[5];
    const float* Wv_p = (const float*)d_in[6];
    const float* bv_p = (const float*)d_in[7];
    const float* Wo_p = (const float*)d_in[8];
    const float* bo_p = (const float*)d_in[9];
    const float* g_p  = (const float*)d_in[10];
    const float* be_p = (const float*)d_in[11];
    const float* Wk_a = (const float*)d_in[12];
    const float* bk_a = (const float*)d_in[13];
    const float* Wq_a = (const float*)d_in[14];
    const float* bq_a = (const float*)d_in[15];
    const float* Wv_a = (const float*)d_in[16];
    const float* bv_a = (const float*)d_in[17];
    const float* Wo_a = (const float*)d_in[18];
    const float* bo_a = (const float*)d_in[19];
    const float* g_a  = (const float*)d_in[20];
    const float* be_a = (const float*)d_in[21];
    const float* R_w  = (const float*)d_in[22];
    const int*   ei_w = (const int*)d_in[23];
    const float* R_c  = (const float*)d_in[24];
    const int*   ei_c = (const int*)d_in[25];
    const float* R_wb = (const float*)d_in[26];
    const int*   ei_wb= (const int*)d_in[27];

    const int N = in_sizes[0] / DIM;
    const int E = in_sizes[23] / 2;
    const size_t NB = (size_t)N * DIM;

    // ---- carve workspace (256B-aligned slabs) ----
    unsigned char* wp = (unsigned char*)d_ws;
    auto take = [&](size_t bytes) -> void* {
        void* p = (void*)wp;
        wp += (bytes + 255) & ~(size_t)255;
        return p;
    };
    unsigned short* xb_p = (unsigned short*)take(NB * 2);
    unsigned short* xb_a = (unsigned short*)take(NB * 2);
    unsigned short* pWk_p = (unsigned short*)take(65536 * 2);
    unsigned short* pWv_p = (unsigned short*)take(65536 * 2);
    unsigned short* pWk_a = (unsigned short*)take(65536 * 2);
    unsigned short* pWv_a = (unsigned short*)take(65536 * 2);
    unsigned short* pWo_p = (unsigned short*)take(65536 * 2);
    unsigned short* pWo_a = (unsigned short*)take(65536 * 2);
    unsigned short* pWq_w = (unsigned short*)take(65536 * 2);
    unsigned short* pWq_c = (unsigned short*)take(65536 * 2);
    unsigned short* pWq_b = (unsigned short*)take(65536 * 2);
    float* weff_w = (float*)take(65536 * 4); float* beff_w = (float*)take(DIM * 4);
    float* weff_c = (float*)take(65536 * 4); float* beff_c = (float*)take(DIM * 4);
    float* weff_b = (float*)take(65536 * 4); float* beff_b = (float*)take(DIM * 4);
    float* Kp = (float*)take(NB * 4);
    float* Vp = (float*)take(NB * 4);
    float* Ka = (float*)take(NB * 4);
    float* Va = (float*)take(NB * 4);
    float* qWw = (float*)take(NB * 4);
    float* qWc = (float*)take(NB * 4);
    float* qWb = (float*)take(NB * 4);
    float* acc_p = (float*)take(NB * 4);
    float* acc_a = (float*)take(NB * 4);
    float* nrm_p = (float*)take((size_t)N * 4);
    float* nrm_a = (float*)take((size_t)N * 4);
    float* y_p = Kp;   // reuse: K/V dead after edge phase
    float* y_a = Ka;

    const int mblocks = (N + 15) / 16;
    const int eblocks = (E + 7) / 8;
    const int lblocks = ((N * 32) + 255) / 256;

    // 1) bf16 conversions
    cvt_bf16_kernel<<<2048, 256, 0, stream>>>(x_p, xb_p, NB);
    cvt_bf16_kernel<<<2048, 256, 0, stream>>>(x_a, xb_a, NB);

    // 2) fold Wq @ R into effective query-projection matrices (one per relation)
    wqeff_kernel<<<257, 256, 0, stream>>>(Wq_p, bq_p, R_w,  weff_w, beff_w);
    wqeff_kernel<<<257, 256, 0, stream>>>(Wq_p, bq_p, R_c,  weff_c, beff_c);
    wqeff_kernel<<<257, 256, 0, stream>>>(Wq_a, bq_a, R_wb, weff_b, beff_b);

    // 3) pack all weight matrices into WMMA B-fragment order (bf16)
    pack_b_kernel<<<256, 256, 0, stream>>>(Wk_p,   pWk_p);
    pack_b_kernel<<<256, 256, 0, stream>>>(Wv_p,   pWv_p);
    pack_b_kernel<<<256, 256, 0, stream>>>(Wk_a,   pWk_a);
    pack_b_kernel<<<256, 256, 0, stream>>>(Wv_a,   pWv_a);
    pack_b_kernel<<<256, 256, 0, stream>>>(Wo_p,   pWo_p);
    pack_b_kernel<<<256, 256, 0, stream>>>(Wo_a,   pWo_a);
    pack_b_kernel<<<256, 256, 0, stream>>>(weff_w, pWq_w);
    pack_b_kernel<<<256, 256, 0, stream>>>(weff_c, pWq_c);
    pack_b_kernel<<<256, 256, 0, stream>>>(weff_b, pWq_b);

    // 4) zero aggregation buffers
    hipMemsetAsync(acc_p, 0, NB * 4, stream);
    hipMemsetAsync(acc_a, 0, NB * 4, stream);
    hipMemsetAsync(nrm_p, 0, (size_t)N * 4, stream);
    hipMemsetAsync(nrm_a, 0, (size_t)N * 4, stream);

    // 5) WMMA projection GEMMs (K, V, and pre-rotated Q per relation)
    gemm_bias_kernel<<<mblocks, 256, 0, stream>>>(xb_p, pWk_p, bk_p,   Kp,  N);
    gemm_bias_kernel<<<mblocks, 256, 0, stream>>>(xb_p, pWv_p, bv_p,   Vp,  N);
    gemm_bias_kernel<<<mblocks, 256, 0, stream>>>(xb_a, pWk_a, bk_a,   Ka,  N);
    gemm_bias_kernel<<<mblocks, 256, 0, stream>>>(xb_a, pWv_a, bv_a,   Va,  N);
    gemm_bias_kernel<<<mblocks, 256, 0, stream>>>(xb_p, pWq_w, beff_w, qWw, N);
    gemm_bias_kernel<<<mblocks, 256, 0, stream>>>(xb_p, pWq_c, beff_c, qWc, N);
    gemm_bias_kernel<<<mblocks, 256, 0, stream>>>(xb_a, pWq_b, beff_b, qWb, N);

    // 6) edge message + scatter (writes: a->p, cites: p->p, writtenby: p->a)
    edge_kernel<<<eblocks, 256, 0, stream>>>(qWw, Ka, Va, ei_w,  E, acc_p, nrm_p);
    edge_kernel<<<eblocks, 256, 0, stream>>>(qWc, Kp, Vp, ei_c,  E, acc_p, nrm_p);
    edge_kernel<<<eblocks, 256, 0, stream>>>(qWb, Kp, Vp, ei_wb, E, acc_a, nrm_a);

    // 7) output projection (normalize + bf16 fragments on the fly) + residual
    gemm_out_kernel<<<mblocks, 256, 0, stream>>>(acc_p, nrm_p, pWo_p, bo_p, x_p, y_p, N);
    gemm_out_kernel<<<mblocks, 256, 0, stream>>>(acc_a, nrm_a, pWo_a, bo_a, x_a, y_a, N);

    // 8) LayerNorm -> stacked output [2, N, 256]
    float* out = (float*)d_out;
    ln_kernel<<<lblocks, 256, 0, stream>>>(y_p, g_p, be_p, out,      N);
    ln_kernel<<<lblocks, 256, 0, stream>>>(y_a, g_a, be_a, out + NB, N);
}
